// Postal_temporal_GIN_GRU_78099685310580
// MI455X (gfx1250) — matrix operand order repaired and checked by reading, not compiled, not run on hardware
//
#include <hip/hip_runtime.h>
#include <cstdint>

// ---- problem constants (match reference) ----
#define T_STEPS 12
#define N_NODES 50000
#define E_EDGES 800000
#define IN_DIM 12
#define H_DIM 128
#define H2_DIM 64
#define H3_DIM 32
#define B_ROWS 4096
#define AF_DIM 10
#define BN_EPS 1e-5f

// N_NODES/16 = 3125 row tiles, 4 waves (tiles) per block
#define ROW_TILES 3125
#define ROW_BLOCKS 782   // ceil(3125/4)

typedef _Float16 half_t;
typedef __attribute__((ext_vector_type(16))) _Float16 v16h;
typedef __attribute__((ext_vector_type(8)))  _Float16 v8h;
typedef __attribute__((ext_vector_type(8)))  float    v8f;

// ---------------------------------------------------------------------------
// WMMA operand loaders (wave32, V_WMMA_F32_16X16X32_F16 layouts per ISA 7.12.2)
//   A 16x32 f16: lane l -> row M = l&15; K-half = l>>4; slots 0..7 -> K = 8*half+0..7,
//                slots 8..15 -> K = 16 + 8*half + 0..7   (two contiguous 8-float runs)
//   B 32x16 f16: lane l -> col N = l&15; slots j -> K = 8*(l>>4) + (j&7) + 16*(j>>3) (pre-packed)
//   C/D 16x16 f32: lane l -> col N = l&15; vreg r -> row M = r + 8*(l>>4)
// ---------------------------------------------------------------------------
__device__ __forceinline__ v16h load_A_global(const float* __restrict__ X, int ld,
                                              int row0, int k0, int lane) {
  const float* p = X + (size_t)(row0 + (lane & 15)) * ld + k0 + ((lane >> 4) << 3);
  // two contiguous 8-float runs -> four b128 loads, packed f32->f16 converts
  const float4 f0 = *(const float4*)(p);
  const float4 f1 = *(const float4*)(p + 4);
  const float4 f2 = *(const float4*)(p + 16);
  const float4 f3 = *(const float4*)(p + 20);
  v16h a;
  a[0]  = (_Float16)f0.x; a[1]  = (_Float16)f0.y; a[2]  = (_Float16)f0.z; a[3]  = (_Float16)f0.w;
  a[4]  = (_Float16)f1.x; a[5]  = (_Float16)f1.y; a[6]  = (_Float16)f1.z; a[7]  = (_Float16)f1.w;
  a[8]  = (_Float16)f2.x; a[9]  = (_Float16)f2.y; a[10] = (_Float16)f2.z; a[11] = (_Float16)f2.w;
  a[12] = (_Float16)f3.x; a[13] = (_Float16)f3.y; a[14] = (_Float16)f3.z; a[15] = (_Float16)f3.w;
  return a;
}

__device__ __forceinline__ v16h load_A_lds(const half_t* S, int ld, int k0, int lane) {
  const half_t* p = S + (lane & 15) * ld + k0 + ((lane >> 4) << 3);
  const v8h x0 = *(const v8h*)(p);        // 16-byte LDS loads
  const v8h x1 = *(const v8h*)(p + 16);
  v16h a;
#pragma unroll
  for (int j = 0; j < 8; ++j) { a[j] = x0[j]; a[j + 8] = x1[j]; }
  return a;
}

__device__ __forceinline__ v16h load_B_packed(const half_t* __restrict__ Wpk,
                                              int kt, int nt, int nKt, int lane) {
  return *(const v16h*)(Wpk + ((((size_t)nt * nKt + kt) * 32 + lane) << 4));
}

// ---------------------------------------------------------------------------
// Pack torch Linear weight W[M,K] (row-major) into wave32 B-operand tiles of
// W^T (K x M), zero-padding K up to Kpad. Packed index == flat thread index.
// ---------------------------------------------------------------------------
__global__ void pack_weight_kernel(const float* __restrict__ W, half_t* __restrict__ Wpk,
                                   int K, int M, int Kpad) {
  int nKt = Kpad >> 5;
  long total = (long)(M >> 4) * nKt * 32 * 16;
  for (long idx = blockIdx.x * (long)blockDim.x + threadIdx.x; idx < total;
       idx += (long)gridDim.x * blockDim.x) {
    int j    = (int)(idx & 15);
    int lane = (int)((idx >> 4) & 31);
    long tile = idx >> 9;
    int kt = (int)(tile % nKt);
    int nt = (int)(tile / nKt);
    int n = nt * 16 + (lane & 15);
    int k = kt * 32 + ((lane >> 4) << 3) + (j & 7) + ((j >> 3) << 4);
    float v = (k < K) ? W[(size_t)n * K + k] : 0.0f;
    Wpk[idx] = (half_t)v;
  }
}

// Precompute BN affine: scale = g*rsqrt(rv+eps), shift = be - rm*scale
__global__ void bn_prep_kernel(const float* __restrict__ g, const float* __restrict__ be,
                               const float* __restrict__ rm, const float* __restrict__ rv,
                               float* __restrict__ sc, float* __restrict__ sh) {
  int i = threadIdx.x;
  if (i < H_DIM) {
    float s = g[i] * rsqrtf(rv[i] + BN_EPS);
    sc[i] = s;
    sh[i] = be[i] - rm[i] * s;
  }
}

__global__ void zero_kernel(float* __restrict__ p, long n) {
  for (long i = blockIdx.x * (long)blockDim.x + threadIdx.x; i < n;
       i += (long)gridDim.x * blockDim.x) p[i] = 0.0f;
}

// z1[N,32] = [ x_t[N,12] | zeros ]  (K padded to 32 for WMMA)
__global__ void build_z1_kernel(const float* __restrict__ x_t, float* __restrict__ z1) {
  long total = (long)N_NODES * 32;
  for (long idx = blockIdx.x * (long)blockDim.x + threadIdx.x; idx < total;
       idx += (long)gridDim.x * blockDim.x) {
    int i = (int)(idx >> 5), c = (int)(idx & 31);
    z1[idx] = (c < IN_DIM) ? x_t[(size_t)i * IN_DIM + c] : 0.0f;
  }
}

__global__ void copy_kernel(const float* __restrict__ s, float* __restrict__ d, long n) {
  for (long i = blockIdx.x * (long)blockDim.x + threadIdx.x; i < n;
       i += (long)gridDim.x * blockDim.x) d[i] = s[i];
}

// GIN aggregation, 12-wide: z1[dst,c] += x_t[src,c]   (L2-resident atomics)
__global__ void scatter12_kernel(const float* __restrict__ x_t,
                                 const int* __restrict__ src, const int* __restrict__ dst,
                                 float* __restrict__ z1) {
  long total = (long)E_EDGES * IN_DIM;
  for (long idx = blockIdx.x * (long)blockDim.x + threadIdx.x; idx < total;
       idx += (long)gridDim.x * blockDim.x) {
    int e = (int)(idx / IN_DIM), c = (int)(idx % IN_DIM);
    atomicAdd(&z1[(size_t)dst[e] * 32 + c], x_t[(size_t)src[e] * IN_DIM + c]);
  }
}

// GIN aggregation, 128-wide: z2[dst,4c..] += h[src,4c..] (float4 gather, 4 atomics)
__global__ void scatter128_kernel(const float* __restrict__ h,
                                  const int* __restrict__ src, const int* __restrict__ dst,
                                  float* __restrict__ z2) {
  long total = (long)E_EDGES * 32;
  for (long idx = blockIdx.x * (long)blockDim.x + threadIdx.x; idx < total;
       idx += (long)gridDim.x * blockDim.x) {
    int e = (int)(idx >> 5), q = (int)(idx & 31);
    const float4 v = *(const float4*)(h + (size_t)src[e] * H_DIM + q * 4);
    float* o = z2 + (size_t)dst[e] * H_DIM + q * 4;
    atomicAdd(o + 0, v.x); atomicAdd(o + 1, v.y);
    atomicAdd(o + 2, v.z); atomicAdd(o + 3, v.w);
  }
}

// ---------------------------------------------------------------------------
// Fused GIN MLP: Out = relu( bn(relu(Z@W1^T+b1)) @ W2^T + b2 ), per-wave 16-row
// tile; layer-1 result staged in LDS as f16 (per-wave private, in-order DS).
// ---------------------------------------------------------------------------
__global__ void gin_mlp_kernel(const float* __restrict__ Z, int ldz, int Kpad,
                               const half_t* __restrict__ W1pk, const float* __restrict__ b1,
                               const float* __restrict__ bnsc, const float* __restrict__ bnsh,
                               const half_t* __restrict__ W2pk, const float* __restrict__ b2,
                               float* __restrict__ OutE, int nrows) {
  __shared__ half_t hmid[4][16 * H_DIM];  // 16 KB
  int wave = threadIdx.x >> 5, lane = threadIdx.x & 31;
  int rt = blockIdx.x * 4 + wave;
  if (rt * 16 >= nrows) return;
  int row0 = rt * 16;
  int nKt1 = Kpad >> 5;
  half_t* S = hmid[wave];
  int ncol = lane & 15, rbase = (lane >> 4) << 3;

  // ---- layer 1: [16,Kpad] @ [Kpad,128] ----
  v8f acc[8];
#pragma unroll
  for (int nt = 0; nt < 8; ++nt) {
    float bb = b1[nt * 16 + ncol];
#pragma unroll
    for (int r = 0; r < 8; ++r) acc[nt][r] = bb;
  }
  for (int kt = 0; kt < nKt1; ++kt) {
    v16h a = load_A_global(Z, ldz, row0, kt << 5, lane);
#pragma unroll
    for (int nt = 0; nt < 8; ++nt) {
      v16h b = load_B_packed(W1pk, kt, nt, nKt1, lane);
      acc[nt] = __builtin_amdgcn_wmma_f32_16x16x32_f16(false, a, false, b,
                                                       (short)0, acc[nt], false, false);
    }
  }
#pragma unroll
  for (int nt = 0; nt < 8; ++nt) {
    int n = nt * 16 + ncol;
    float sc = bnsc[n], sh = bnsh[n];
#pragma unroll
    for (int r = 0; r < 8; ++r) {
      float v = fmaxf(acc[nt][r], 0.0f) * sc + sh;
      S[(r + rbase) * H_DIM + n] = (half_t)v;
    }
  }

  // ---- layer 2: [16,128] @ [128,128] ----
#pragma unroll
  for (int nt = 0; nt < 8; ++nt) {
    float bb = b2[nt * 16 + ncol];
#pragma unroll
    for (int r = 0; r < 8; ++r) acc[nt][r] = bb;
  }
#pragma unroll
  for (int kt = 0; kt < 4; ++kt) {
    v16h a = load_A_lds(S, H_DIM, kt << 5, lane);
#pragma unroll
    for (int nt = 0; nt < 8; ++nt) {
      v16h b = load_B_packed(W2pk, kt, nt, 4, lane);
      acc[nt] = __builtin_amdgcn_wmma_f32_16x16x32_f16(false, a, false, b,
                                                       (short)0, acc[nt], false, false);
    }
  }
#pragma unroll
  for (int nt = 0; nt < 8; ++nt) {
    int n = nt * 16 + ncol;
#pragma unroll
    for (int r = 0; r < 8; ++r)
      OutE[(size_t)(row0 + r + rbase) * H_DIM + n] = fmaxf(acc[nt][r], 0.0f);
  }
}

// ---------------------------------------------------------------------------
// Y[N,M] = X[N,128] @ Wpk + bias   (GRU gate GEMMs, M=384); wave = 16x64 strip
// ---------------------------------------------------------------------------
__global__ void wmma_gemm_bias_kernel(const float* __restrict__ X,
                                      const half_t* __restrict__ Wpk,
                                      const float* __restrict__ bias,
                                      float* __restrict__ Y,
                                      int nrows, int K, int M) {
  int wave = threadIdx.x >> 5, lane = threadIdx.x & 31;
  int rt = blockIdx.x * 4 + wave;
  if (rt * 16 >= nrows) return;
  int row0 = rt * 16;
  int c0 = blockIdx.y * 64;
  int nKt = K >> 5;
  int ncol = lane & 15, rbase = (lane >> 4) << 3;
  v8f acc[4];
#pragma unroll
  for (int tt = 0; tt < 4; ++tt) {
    float bb = bias[c0 + tt * 16 + ncol];
#pragma unroll
    for (int r = 0; r < 8; ++r) acc[tt][r] = bb;
  }
  for (int kt = 0; kt < nKt; ++kt) {
    v16h a = load_A_global(X, K, row0, kt << 5, lane);
#pragma unroll
    for (int tt = 0; tt < 4; ++tt) {
      v16h b = load_B_packed(Wpk, kt, (c0 >> 4) + tt, nKt, lane);
      acc[tt] = __builtin_amdgcn_wmma_f32_16x16x32_f16(false, a, false, b,
                                                       (short)0, acc[tt], false, false);
    }
  }
#pragma unroll
  for (int tt = 0; tt < 4; ++tt) {
    int n = c0 + tt * 16 + ncol;
#pragma unroll
    for (int r = 0; r < 8; ++r)
      Y[(size_t)(row0 + r + rbase) * M + n] = acc[tt][r];
  }
}

__device__ __forceinline__ float sigmoidf_(float x) { return 1.0f / (1.0f + __expf(-x)); }

// GRU pointwise gates, h updated in place (gi/gh include biases)
__global__ void gru_gate_kernel(const float* __restrict__ gi, const float* __restrict__ gh,
                                float* __restrict__ h) {
  long total = (long)N_NODES * H_DIM;
  for (long idx = blockIdx.x * (long)blockDim.x + threadIdx.x; idx < total;
       idx += (long)gridDim.x * blockDim.x) {
    int row = (int)(idx >> 7), c = (int)(idx & 127);
    const float* gir = gi + (size_t)row * 384;
    const float* ghr = gh + (size_t)row * 384;
    float r = sigmoidf_(gir[c] + ghr[c]);
    float z = sigmoidf_(gir[128 + c] + ghr[128 + c]);
    float n = tanhf(gir[256 + c] + r * ghr[256 + c]);
    h[idx] = (1.0f - z) * n + z * h[idx];
  }
}

// Head: gather -> fc1(leaky 0.1) -> fc2(leaky 0.05) -> fc3
__global__ void head_kernel(const float* __restrict__ h2, const float* __restrict__ apart,
                            const int* __restrict__ tidx,
                            const float* __restrict__ fc1w, const float* __restrict__ fc1b,
                            const float* __restrict__ fc2w, const float* __restrict__ fc2b,
                            const float* __restrict__ fc3w, const float* __restrict__ fc3b,
                            float* __restrict__ out) {
  int b = blockIdx.x * blockDim.x + threadIdx.x;
  if (b >= B_ROWS) return;
  const float* hrow = h2 + (size_t)tidx[b] * H_DIM;
  const float* arow = apart + (size_t)b * AF_DIM;
  float v1[H2_DIM];
  for (int o = 0; o < H2_DIM; ++o) {
    const float* w = fc1w + (size_t)o * (H_DIM + AF_DIM);
    float s = fc1b[o];
    for (int i = 0; i < H_DIM; ++i) s += w[i] * hrow[i];
    for (int i = 0; i < AF_DIM; ++i) s += w[H_DIM + i] * arow[i];
    v1[o] = (s > 0.0f) ? s : 0.1f * s;
  }
  float v2[H3_DIM];
  for (int o = 0; o < H3_DIM; ++o) {
    const float* w = fc2w + (size_t)o * H2_DIM;
    float s = fc2b[o];
    for (int i = 0; i < H2_DIM; ++i) s += w[i] * v1[i];
    v2[o] = (s > 0.0f) ? s : 0.05f * s;
  }
  float s = fc3b[0];
  for (int i = 0; i < H3_DIM; ++i) s += fc3w[i] * v2[i];
  out[b] = s;
}

// ---------------------------------------------------------------------------
extern "C" void kernel_launch(void* const* d_in, const int* in_sizes, int n_in,
                              void* d_out, int out_size, void* d_ws, size_t ws_size,
                              hipStream_t stream) {
  (void)in_sizes; (void)n_in; (void)out_size; (void)ws_size;
  // inputs in setup_inputs() dict order
  const float* x_seq   = (const float*)d_in[0];
  const float* apart   = (const float*)d_in[1];
  const int*   e_src   = (const int*)d_in[2];
  const int*   e_dst   = (const int*)d_in[3];
  const int*   t_idx   = (const int*)d_in[4];
  const float* c1_w1   = (const float*)d_in[5];
  const float* c1_b1   = (const float*)d_in[6];
  const float* c1_g    = (const float*)d_in[7];
  const float* c1_be   = (const float*)d_in[8];
  const float* c1_rm   = (const float*)d_in[9];
  const float* c1_rv   = (const float*)d_in[10];
  const float* c1_w2   = (const float*)d_in[11];
  const float* c1_b2   = (const float*)d_in[12];
  const float* c2_w1   = (const float*)d_in[13];
  const float* c2_b1   = (const float*)d_in[14];
  const float* c2_g    = (const float*)d_in[15];
  const float* c2_be   = (const float*)d_in[16];
  const float* c2_rm   = (const float*)d_in[17];
  const float* c2_rv   = (const float*)d_in[18];
  const float* c2_w2   = (const float*)d_in[19];
  const float* c2_b2   = (const float*)d_in[20];
  const float* gru_wih = (const float*)d_in[21];
  const float* gru_whh = (const float*)d_in[22];
  const float* gru_bih = (const float*)d_in[23];
  const float* gru_bhh = (const float*)d_in[24];
  const float* fc1_w   = (const float*)d_in[25];
  const float* fc1_b   = (const float*)d_in[26];
  const float* fc2_w   = (const float*)d_in[27];
  const float* fc2_b   = (const float*)d_in[28];
  const float* fc3_w   = (const float*)d_in[29];
  const float* fc3_b   = (const float*)d_in[30];

  // workspace carve-up (~262 MB total)
  char* ws = (char*)d_ws;
  size_t off = 0;
  auto carve = [&](size_t bytes) -> void* {
    void* p = ws + off;
    off = (off + bytes + 255) & ~(size_t)255;
    return p;
  };
  half_t* pk_c1w1 = (half_t*)carve((size_t)32 * 128 * 2);
  half_t* pk_c1w2 = (half_t*)carve((size_t)128 * 128 * 2);
  half_t* pk_c2w1 = (half_t*)carve((size_t)128 * 128 * 2);
  half_t* pk_c2w2 = (half_t*)carve((size_t)128 * 128 * 2);
  half_t* pk_wih  = (half_t*)carve((size_t)128 * 384 * 2);
  half_t* pk_whh  = (half_t*)carve((size_t)128 * 384 * 2);
  float* bn1sc = (float*)carve(128 * 4);
  float* bn1sh = (float*)carve(128 * 4);
  float* bn2sc = (float*)carve(128 * 4);
  float* bn2sh = (float*)carve(128 * 4);
  float* z1 = (float*)carve((size_t)N_NODES * 32 * 4);
  float* z2 = (float*)carve((size_t)N_NODES * H_DIM * 4);
  float* eb = (float*)carve((size_t)N_NODES * H_DIM * 4);
  float* h1 = (float*)carve((size_t)N_NODES * H_DIM * 4);
  float* h2 = (float*)carve((size_t)N_NODES * H_DIM * 4);
  float* gi = (float*)carve((size_t)N_NODES * 384 * 4);
  float* gh = (float*)carve((size_t)N_NODES * 384 * 4);

  // ---- one-time prep ----
  pack_weight_kernel<<<64, 256, 0, stream>>>(c1_w1, pk_c1w1, IN_DIM, H_DIM, 32);
  pack_weight_kernel<<<64, 256, 0, stream>>>(c1_w2, pk_c1w2, H_DIM, H_DIM, 128);
  pack_weight_kernel<<<64, 256, 0, stream>>>(c2_w1, pk_c2w1, H_DIM, H_DIM, 128);
  pack_weight_kernel<<<64, 256, 0, stream>>>(c2_w2, pk_c2w2, H_DIM, H_DIM, 128);
  pack_weight_kernel<<<128, 256, 0, stream>>>(gru_wih, pk_wih, H_DIM, 3 * H_DIM, 128);
  pack_weight_kernel<<<128, 256, 0, stream>>>(gru_whh, pk_whh, H_DIM, 3 * H_DIM, 128);
  bn_prep_kernel<<<1, 128, 0, stream>>>(c1_g, c1_be, c1_rm, c1_rv, bn1sc, bn1sh);
  bn_prep_kernel<<<1, 128, 0, stream>>>(c2_g, c2_be, c2_rm, c2_rv, bn2sc, bn2sh);
  zero_kernel<<<4096, 256, 0, stream>>>(h1, (long)N_NODES * H_DIM);
  zero_kernel<<<4096, 256, 0, stream>>>(h2, (long)N_NODES * H_DIM);

  const dim3 gemmGrid(ROW_BLOCKS, (3 * H_DIM) / 64);  // (782, 6)

  // ---- fully-fused time loop: conv1 -> GRU1 step -> conv2 -> GRU2 step ----
  for (int t = 0; t < T_STEPS; ++t) {
    const float* x_t = x_seq + (size_t)t * N_NODES * IN_DIM;

    // conv1: z1 = [x_t | 0], += neighbor sum, then fused MLP -> eb
    build_z1_kernel<<<6250, 256, 0, stream>>>(x_t, z1);
    scatter12_kernel<<<18750, 256, 0, stream>>>(x_t, e_src, e_dst, z1);
    gin_mlp_kernel<<<ROW_BLOCKS, 128, 0, stream>>>(z1, 32, 32, pk_c1w1, c1_b1,
                                                   bn1sc, bn1sh, pk_c1w2, c1_b2,
                                                   eb, N_NODES);
    // GRU1 step (shared weights)
    wmma_gemm_bias_kernel<<<gemmGrid, 128, 0, stream>>>(eb, pk_wih, gru_bih, gi,
                                                        N_NODES, H_DIM, 3 * H_DIM);
    wmma_gemm_bias_kernel<<<gemmGrid, 128, 0, stream>>>(h1, pk_whh, gru_bhh, gh,
                                                        N_NODES, H_DIM, 3 * H_DIM);
    gru_gate_kernel<<<12500, 256, 0, stream>>>(gi, gh, h1);

    // conv2 on GRU1 output
    copy_kernel<<<12500, 256, 0, stream>>>(h1, z2, (long)N_NODES * H_DIM);
    scatter128_kernel<<<50000, 256, 0, stream>>>(h1, e_src, e_dst, z2);
    gin_mlp_kernel<<<ROW_BLOCKS, 128, 0, stream>>>(z2, H_DIM, 128, pk_c2w1, c2_b1,
                                                   bn2sc, bn2sh, pk_c2w2, c2_b2,
                                                   eb, N_NODES);
    // GRU2 step (same weights, separate state)
    wmma_gemm_bias_kernel<<<gemmGrid, 128, 0, stream>>>(eb, pk_wih, gru_bih, gi,
                                                        N_NODES, H_DIM, 3 * H_DIM);
    wmma_gemm_bias_kernel<<<gemmGrid, 128, 0, stream>>>(h2, pk_whh, gru_bhh, gh,
                                                        N_NODES, H_DIM, 3 * H_DIM);
    gru_gate_kernel<<<12500, 256, 0, stream>>>(gi, gh, h2);
  }

  // ---- head ----
  head_kernel<<<(B_ROWS + 255) / 256, 256, 0, stream>>>(
      h2, apart, t_idx, fc1_w, fc1_b, fc2_w, fc2_b, fc3_w, fc3_b, (float*)d_out);
}
